// NeRF_84164179132769
// MI455X (gfx1250) — matrix operand order repaired and checked by reading, not compiled, not run on hardware
//
#include <hip/hip_runtime.h>

typedef __attribute__((ext_vector_type(16))) _Float16 v16h;
typedef __attribute__((ext_vector_type(8)))  _Float16 v8h;
typedef __attribute__((ext_vector_type(8)))  float    v8f;

#define BLOCK 256

__device__ __forceinline__ v8f wmma16(v16h a, v16h b, v8f c) {
  // D = A(16x32 f16) * B(32x16 f16) + C(16x16 f32)
  return __builtin_amdgcn_wmma_f32_16x16x32_f16(false, a, false, b, (short)0, c, false, false);
}

__device__ __forceinline__ v8h ld8(const _Float16* p) { return *(const v8h*)p; }

__device__ __forceinline__ v16h cat16(v8h lo, v8h hi) {
  return __builtin_shufflevector(lo, hi, 0,1,2,3,4,5,6,7,8,9,10,11,12,13,14,15);
}

// 16 contiguous halves (B fragments: Wt[n][kpad], k contiguous per lane)
__device__ __forceinline__ v16h ld16(const _Float16* p) { return cat16(ld8(p), ld8(p + 8)); }

// ---- one hidden layer (32 outputs, ReLU) for both 16-point tiles ----
// aB = feat + (lane&15)*64 + hi*16         (A frag base; +t*1024, halves at +0/+8, kblock at +32/+40)
// sB = feat + (lane&15)   + hi*512         (store base; element (t,v) at +(t*16+v)*64, ntile1 at +16)
// bB = wl   + (lane&15)*KP + hi*16         (B frag base; ntile1 at +16*KP, kblock1 at +32)
template<int KB>
__device__ __forceinline__ void layer32(const _Float16* aB, _Float16* sB,
                                        const _Float16* bB, const float* bl, int lane) {
  constexpr int KP = KB * 32;
  const v16h b00 = ld16(bB);
  const v16h b10 = ld16(bB + 16 * KP);
  v16h b01, b11;
  if constexpr (KB == 2) {
    b01 = ld16(bB + 32);
    b11 = ld16(bB + 16 * KP + 32);
  }
  v8f acc[2][2];
#pragma unroll
  for (int t = 0; t < 2; ++t) {
    const v8f z = {};
    const v16h a0 = cat16(ld8(aB + t * 1024), ld8(aB + t * 1024 + 8));
    acc[t][0] = wmma16(a0, b00, z);
    acc[t][1] = wmma16(a0, b10, z);
    if constexpr (KB == 2) {
      const v16h a1 = cat16(ld8(aB + t * 1024 + 32), ld8(aB + t * 1024 + 40));
      acc[t][0] = wmma16(a1, b01, acc[t][0]);
      acc[t][1] = wmma16(a1, b11, acc[t][1]);
    }
  }
  const int n = lane & 15;
  const float bn0 = bl[n], bn1 = bl[n + 16];
  asm volatile("" ::: "memory");   // all A loads issued before overwriting feat
#pragma unroll
  for (int t = 0; t < 2; ++t) {
#pragma unroll
    for (int v = 0; v < 8; ++v) {
      const float x0 = fmaxf(acc[t][0][v] + bn0, 0.0f);
      const float x1 = fmaxf(acc[t][1][v] + bn1, 0.0f);
      sB[(t * 16 + v) * 64]      = (_Float16)x0;   // immediate DS offsets
      sB[(t * 16 + v) * 64 + 16] = (_Float16)x1;
    }
  }
  asm volatile("" ::: "memory");
}

// ---- final layer: 32 -> ncols (N padded to 16); writes f32 to out[32][4] ----
__device__ __forceinline__ void final_layer(const _Float16* aB, const _Float16* bB,
                                            const float* bl, int lane,
                                            float* outw, int colbase, int ncols) {
  const v16h b = ld16(bB);
  v8f c[2];
#pragma unroll
  for (int t = 0; t < 2; ++t) {
    const v8f z = {};
    const v16h a = cat16(ld8(aB + t * 1024), ld8(aB + t * 1024 + 8));
    c[t] = wmma16(a, b, z);
  }
  const int n  = lane & 15;
  const int hi = (lane >> 4) & 1;
  if (n < ncols) {
    const float bn = bl[n];
    float* oB = outw + hi * 32 + colbase + n;   // m = t*16 + v + 8*hi -> m*4 + col
#pragma unroll
    for (int t = 0; t < 2; ++t) {
#pragma unroll
      for (int v = 0; v < 8; ++v) {
        oB[(t * 16 + v) * 4] = c[t][v] + bn;
      }
    }
  }
  asm volatile("" ::: "memory");
}

// zero this lane's feat row and write raw inputs at k=0.. and the skip copy at k=32..
__device__ __forceinline__ void write_inputs(_Float16* row, const float* vals, int cnt) {
  v8h z = {};
#pragma unroll
  for (int i = 0; i < 8; ++i) *(v8h*)(row + i * 8) = z;
#pragma unroll
  for (int i = 0; i < 7; ++i) {
    if (i < cnt) {
      const _Float16 h = (_Float16)vals[i];
      row[i]      = h;
      row[32 + i] = h;
    }
  }
  asm volatile("" ::: "memory");
}

// ---- trilinear sample (faithful grid_sample align_corners=True, zero padding) ----
template<int D, int H, int W, bool RGB>
__device__ __forceinline__ void sample_level(const float* __restrict__ g,
                                             float px, float py, float pz, float glen,
                                             float* __restrict__ alpha,
                                             float* __restrict__ rgb) {
  const float xs = (px + 4.0f) / ((float)W * glen) * 2.0f - 1.0f;
  const float ys = (py + 4.0f) / ((float)H * glen) * 2.0f - 1.0f;
  const float zs = (pz + 2.0f) / ((float)D * glen) * 2.0f - 1.0f;
  const float ix = (xs + 1.0f) * 0.5f * (float)(W - 1);
  const float iy = (ys + 1.0f) * 0.5f * (float)(H - 1);
  const float iz = (zs + 1.0f) * 0.5f * (float)(D - 1);
  const float xf = floorf(ix), yf = floorf(iy), zf = floorf(iz);
  const int x0 = (int)xf, y0 = (int)yf, z0 = (int)zf;
  const float fx = ix - xf, fy = iy - yf, fz = iz - zf;
  const float wxv[2] = {1.0f - fx, fx};
  const float wyv[2] = {1.0f - fy, fy};
  const float wzv[2] = {1.0f - fz, fz};
  constexpr int HW  = H * W;
  constexpr int DHW = D * H * W;
  float a = 0.0f, r0 = 0.0f, r1 = 0.0f, r2 = 0.0f;
#pragma unroll
  for (int dz = 0; dz < 2; ++dz) {
#pragma unroll
    for (int dy = 0; dy < 2; ++dy) {
#pragma unroll
      for (int dx = 0; dx < 2; ++dx) {
        const int X = x0 + dx, Y = y0 + dy, Z = z0 + dz;
        const bool ok = (X >= 0) && (X < W) && (Y >= 0) && (Y < H) && (Z >= 0) && (Z < D);
        const int Xc = X < 0 ? 0 : (X > W - 1 ? W - 1 : X);
        const int Yc = Y < 0 ? 0 : (Y > H - 1 ? H - 1 : Y);
        const int Zc = Z < 0 ? 0 : (Z > D - 1 ? D - 1 : Z);
        float w = wzv[dz] * wyv[dy] * wxv[dx];
        w = ok ? w : 0.0f;
        const int off = Zc * HW + Yc * W + Xc;
        a += w * g[3 * DHW + off];
        if constexpr (RGB) {
          r0 += w * g[off];
          r1 += w * g[DHW + off];
          r2 += w * g[2 * DHW + off];
        }
      }
    }
  }
  *alpha = a;
  if constexpr (RGB) { rgb[0] = r0; rgb[1] = r1; rgb[2] = r2; }
}

// ---- weight/bias staging into LDS (transposed, f16, zero-padded) ----
__device__ __forceinline__ void stage_w(const float* __restrict__ w, _Float16* dst,
                                        int fi, int fo, int kpad, int npad, int tid) {
  for (int t = tid; t < npad * kpad; t += BLOCK) {
    const int n = t / kpad;
    const int k = t - n * kpad;
    const float v = (k < fi && n < fo) ? w[k * fo + n] : 0.0f;
    dst[n * kpad + k] = (_Float16)v;
  }
}
__device__ __forceinline__ void stage_b(const float* __restrict__ b, float* dst,
                                        int fo, int npad, int tid) {
  for (int t = tid; t < npad; t += BLOCK) dst[t] = (t < fo) ? b[t] : 0.0f;
}

__global__ __launch_bounds__(BLOCK) void nerf_fused(
    const float* __restrict__ p,
    const float* __restrict__ g0, const float* __restrict__ g1,
    const float* __restrict__ g2, const float* __restrict__ g3,
    const float* __restrict__ g4, const float* __restrict__ g5,
    const float* __restrict__ g6,
    const float* aw0, const float* ab0, const float* aw1, const float* ab1,
    const float* aw2, const float* ab2, const float* aw3, const float* ab3,
    const float* aw4, const float* ab4,
    const float* cw0, const float* cb0, const float* cw1, const float* cb1,
    const float* cw2, const float* cb2, const float* cw3, const float* cb3,
    const float* cw4, const float* cb4,
    float* __restrict__ out, int npts) {
  __shared__ alignas(16) _Float16 s_wa[5632];          // 1024+1024+2048+1024+512
  __shared__ alignas(16) _Float16 s_wc[5632];
  __shared__ float s_ba[160];
  __shared__ float s_bc[160];
  __shared__ alignas(16) _Float16 s_feat[8][32][64];
  __shared__ alignas(16) float    s_out[8][32][4];

  const int tid = threadIdx.x;

  stage_w(aw0, s_wa + 0,     7, 32, 32, 32, tid);
  stage_w(aw1, s_wa + 1024, 32, 32, 32, 32, tid);
  stage_w(aw2, s_wa + 2048, 39, 32, 64, 32, tid);
  stage_w(aw3, s_wa + 4096, 32, 32, 32, 32, tid);
  stage_w(aw4, s_wa + 5120, 32,  1, 32, 16, tid);
  stage_w(cw0, s_wc + 0,     6, 32, 32, 32, tid);
  stage_w(cw1, s_wc + 1024, 32, 32, 32, 32, tid);
  stage_w(cw2, s_wc + 2048, 38, 32, 64, 32, tid);
  stage_w(cw3, s_wc + 4096, 32, 32, 32, 32, tid);
  stage_w(cw4, s_wc + 5120, 32,  3, 32, 16, tid);
  stage_b(ab0, s_ba + 0,   32, 32, tid);
  stage_b(ab1, s_ba + 32,  32, 32, tid);
  stage_b(ab2, s_ba + 64,  32, 32, tid);
  stage_b(ab3, s_ba + 96,  32, 32, tid);
  stage_b(ab4, s_ba + 128,  1, 16, tid);
  stage_b(cb0, s_bc + 0,   32, 32, tid);
  stage_b(cb1, s_bc + 32,  32, 32, tid);
  stage_b(cb2, s_bc + 64,  32, 32, tid);
  stage_b(cb3, s_bc + 96,  32, 32, tid);
  stage_b(cb4, s_bc + 128,  3, 16, tid);
  __syncthreads();

  const int lane = tid & 31;
  const int wave = tid >> 5;
  const int pt   = blockIdx.x * BLOCK + tid;
  const int ptc  = pt < npts ? pt : npts - 1;   // clamp loads; keep EXEC full for WMMA

  const float px = p[ptc * 3 + 0];
  const float py = p[ptc * 3 + 1];
  const float pz = p[ptc * 3 + 2];

  float af[7];
  float rf[6];
  sample_level< 6, 12, 12, false>(g0, px, py, pz, 0.64f, &af[0], nullptr);
  sample_level< 8, 16, 16, false>(g1, px, py, pz, 0.48f, &af[1], nullptr);
  sample_level<12, 25, 25, false>(g2, px, py, pz, 0.32f, &af[2], nullptr);
  sample_level<16, 33, 33, false>(g3, px, py, pz, 0.24f, &af[3], nullptr);
  sample_level<25, 50, 50, false>(g4, px, py, pz, 0.16f, &af[4], nullptr);
  sample_level<33, 66, 66, true >(g5, px, py, pz, 0.12f, &af[5], &rf[0]);
  sample_level<50,100,100, true >(g6, px, py, pz, 0.08f, &af[6], &rf[3]);

  _Float16* feat = &s_feat[wave][0][0];
  float*    outw = &s_out[wave][0][0];

  // lane-invariant bases; all tile/kblock/ntile displacements are immediates
  const int nn = lane & 15;
  const int hi = (lane >> 4) & 1;
  const _Float16* aB = feat + nn * 64 + hi * 16;
  _Float16*       sB = feat + nn + hi * 512;

  // ---------------- alpha MLP: 7 -> 32 -> 32 -> (32+7) -> 32 -> 1 ----------------
  write_inputs(feat + lane * 64, af, 7);
  layer32<1>(aB, sB, s_wa + 0    + nn * 32 + hi * 16, s_ba + 0,  lane);
  layer32<1>(aB, sB, s_wa + 1024 + nn * 32 + hi * 16, s_ba + 32, lane);
  layer32<2>(aB, sB, s_wa + 2048 + nn * 64 + hi * 16, s_ba + 64, lane);
  layer32<1>(aB, sB, s_wa + 4096 + nn * 32 + hi * 16, s_ba + 96, lane);
  final_layer(aB, s_wa + 5120 + nn * 32 + hi * 16, s_ba + 128, lane, outw, 3, 1);

  // ---------------- rgb MLP: 6 -> 32 -> 32 -> (32+6) -> 32 -> 3 ----------------
  write_inputs(feat + lane * 64, rf, 6);
  layer32<1>(aB, sB, s_wc + 0    + nn * 32 + hi * 16, s_bc + 0,  lane);
  layer32<1>(aB, sB, s_wc + 1024 + nn * 32 + hi * 16, s_bc + 32, lane);
  layer32<2>(aB, sB, s_wc + 2048 + nn * 64 + hi * 16, s_bc + 64, lane);
  layer32<1>(aB, sB, s_wc + 4096 + nn * 32 + hi * 16, s_bc + 96, lane);
  final_layer(aB, s_wc + 5120 + nn * 32 + hi * 16, s_bc + 128, lane, outw, 0, 3);

  asm volatile("" ::: "memory");
  if (pt < npts) {
    const float4 v = *(const float4*)(outw + lane * 4);
    *(float4*)(out + (size_t)pt * 4) = v;   // [rgb0,rgb1,rgb2,alpha]
  }
}

extern "C" void kernel_launch(void* const* d_in, const int* in_sizes, int n_in,
                              void* d_out, int out_size, void* d_ws, size_t ws_size,
                              hipStream_t stream) {
  (void)n_in; (void)out_size; (void)d_ws; (void)ws_size;
  const float* p  = (const float*)d_in[0];
  const float* g0 = (const float*)d_in[1];
  const float* g1 = (const float*)d_in[2];
  const float* g2 = (const float*)d_in[3];
  const float* g3 = (const float*)d_in[4];
  const float* g4 = (const float*)d_in[5];
  const float* g5 = (const float*)d_in[6];
  const float* g6 = (const float*)d_in[7];
  const float* aw0 = (const float*)d_in[8];   const float* ab0 = (const float*)d_in[9];
  const float* aw1 = (const float*)d_in[10];  const float* ab1 = (const float*)d_in[11];
  const float* aw2 = (const float*)d_in[12];  const float* ab2 = (const float*)d_in[13];
  const float* aw3 = (const float*)d_in[14];  const float* ab3 = (const float*)d_in[15];
  const float* aw4 = (const float*)d_in[16];  const float* ab4 = (const float*)d_in[17];
  const float* cw0 = (const float*)d_in[18];  const float* cb0 = (const float*)d_in[19];
  const float* cw1 = (const float*)d_in[20];  const float* cb1 = (const float*)d_in[21];
  const float* cw2 = (const float*)d_in[22];  const float* cb2 = (const float*)d_in[23];
  const float* cw3 = (const float*)d_in[24];  const float* cb3 = (const float*)d_in[25];
  const float* cw4 = (const float*)d_in[26];  const float* cb4 = (const float*)d_in[27];

  const int npts = in_sizes[0] / 3;
  const int blocks = (npts + BLOCK - 1) / BLOCK;
  nerf_fused<<<blocks, BLOCK, 0, stream>>>(
      p, g0, g1, g2, g3, g4, g5, g6,
      aw0, ab0, aw1, ab1, aw2, ab2, aw3, ab3, aw4, ab4,
      cw0, cb0, cw1, cb1, cw2, cb2, cw3, cb3, cw4, cb4,
      (float*)d_out, npts);
}
